// TriClip_85916525789537
// MI455X (gfx1250) — matrix operand order repaired and checked by reading, not compiled, not run on hardware
//
#include <hip/hip_runtime.h>
#include <hip/hip_bf16.h>

#define B_ 2
#define C_ 64
#define N_ 1024
#define NN2 (N_ * N_)

typedef __attribute__((ext_vector_type(16))) _Float16     v16h;
typedef __attribute__((ext_vector_type(8)))  float        v8f;
typedef __attribute__((ext_vector_type(4)))  unsigned int v4u;

union Frag16 { v16h h; v4u u[2]; };

#if defined(__HIP_DEVICE_COMPILE__) && \
    __has_builtin(__builtin_amdgcn_tensor_load_to_lds)
#define USE_TDM 1
#else
#define USE_TDM 0
#endif

#if USE_TDM
typedef __attribute__((ext_vector_type(4))) unsigned int u32x4;
typedef __attribute__((ext_vector_type(8))) int          i32x8;
typedef __attribute__((ext_vector_type(4))) int          i32x4;

// Issue a TDM DMA of a 2D f16 tile (rows x cols, row stride in elements)
// from global memory into LDS at byte offset lds_off. D# layout per
// CDNA5 ISA ch.8: group0 = {count/type/addr}, group1 = {dims/strides}.
__device__ __forceinline__ void tdm_load_tile_f16(
    const _Float16* gptr, unsigned int lds_off,
    unsigned int rows, unsigned int cols, unsigned int stride_elems)
{
    const unsigned long long ga = (unsigned long long)(uintptr_t)gptr;
    u32x4 g0;
    g0.x = 1u;                                        // count=1, user mode
    g0.y = lds_off;                                   // lds_addr (bytes)
    g0.z = (unsigned int)(ga & 0xFFFFFFFFu);          // global_addr[31:0]
    g0.w = (unsigned int)((ga >> 32) & 0x01FFFFFFu)   // global_addr[56:32]
           | (2u << 30);                              // type = 2 ("image")

    i32x8 g1;
    g1[0] = (int)(1u << 16);                          // data_size=1 (2 bytes)
    g1[1] = (int)((cols & 0xFFFFu) << 16);            // tensor_dim0[15:0]
    g1[2] = (int)(((cols >> 16) & 0xFFFFu)            // tensor_dim0[31:16]
                  | ((rows & 0xFFFFu) << 16));        // tensor_dim1[15:0]
    g1[3] = (int)(((rows >> 16) & 0xFFFFu)            // tensor_dim1[31:16]
                  | ((cols & 0xFFFFu) << 16));        // tile_dim0
    g1[4] = (int)(rows & 0xFFFFu);                    // tile_dim1 (tile2=0)
    g1[5] = (int)stride_elems;                        // dim0_stride[31:0]
    g1[6] = 0;                                        // stride hi / dim1_stride
    g1[7] = 0;
    const i32x4 z4 = {0, 0, 0, 0};
#if __clang_major__ >= 23
    const i32x8 z8 = {0, 0, 0, 0, 0, 0, 0, 0};
    __builtin_amdgcn_tensor_load_to_lds(g0, g1, z4, z4, z8, 0);
#else
    __builtin_amdgcn_tensor_load_to_lds(g0, g1, z4, z4, 0);
#endif
}
#endif  // USE_TDM

// ===================== WMMA GEMM:  C[b] = A[b] * B[b]^T (+C, +bias) ==========
// A: [M x K] row-major f16 (lda), B: [Nn x K] row-major f16 (ldb),
// C: [M x Nn] fp32 (ldc). grid = (Nn/64, M/64, B). block = 128 (4 waves).
// Each wave computes a 16x64 strip with v_wmma_f32_16x16x32_f16; the shared
// 64x32 B-subtile is DMA'd into LDS by the Tensor Data Mover (double
// buffered, TENSORcnt), A fragments stream from global as b128 loads.
// All 4 B fragments are preloaded into distinct registers so the 8
// ds_load_b128 issue as one clause and the 4 WMMAs fire back-to-back.
__global__ __launch_bounds__(128) void wmma_gemm_nt(
    const _Float16* __restrict__ A, int lda, long sA,
    const _Float16* __restrict__ Bm, int ldb, long sB,
    float* __restrict__ Co, int ldc, long sC,
    _Float16* __restrict__ Ch,            // optional f16 mirror of C
    const float* __restrict__ bias,       // optional [Nn]
    int K, int accumulate)
{
    const int lane = threadIdx.x & 31;
    const int wave = threadIdx.x >> 5;
    const int lr = lane & 15;   // row/col within 16-tile
    const int lg = lane >> 4;   // lane group (K-split selector)
    const int bz = blockIdx.z;
    const int i0 = blockIdx.y * 64 + wave * 16;
    const int j0 = blockIdx.x * 64;

    const _Float16* Ab = A + (size_t)bz * sA;
    const _Float16* Bb = Bm + (size_t)bz * sB;

    v8f acc[4] = {};

#if USE_TDM
    __shared__ _Float16 Bs[2][64 * 32];
    const unsigned int ldsB[2] = {
        (unsigned int)(uintptr_t)(void*)&Bs[0][0],
        (unsigned int)(uintptr_t)(void*)&Bs[1][0]};
    const _Float16* Btile = Bb + (size_t)j0 * ldb;

    if (wave == 0) {
        tdm_load_tile_f16(Btile, ldsB[0], 64, 32, (unsigned int)ldb);
        __builtin_amdgcn_s_wait_tensorcnt(0);
    }
    __syncthreads();

    int cur = 0;
    for (int k0 = 0; k0 < K; k0 += 32) {
        if (k0 + 32 < K && wave == 0)   // DMA next subtile while computing
            tdm_load_tile_f16(Btile + (k0 + 32), ldsB[cur ^ 1], 64, 32,
                              (unsigned int)ldb);

        Frag16 af;
        const _Float16* ap = Ab + (size_t)(i0 + lr) * lda + (k0 + lg * 16);
        af.u[0] = *(const v4u*)(ap);
        af.u[1] = *(const v4u*)(ap + 8);
        if (k0 + 32 < K) __builtin_prefetch(ap + 32, 0, 1);

        Frag16 bf[4];
#pragma unroll
        for (int t = 0; t < 4; ++t) {
            const v4u* bp = (const v4u*)&Bs[cur][(t * 16 + lr) * 32 + lg * 16];
            bf[t].u[0] = bp[0];
            bf[t].u[1] = bp[1];
        }
#pragma unroll
        for (int t = 0; t < 4; ++t)
            acc[t] = __builtin_amdgcn_wmma_f32_16x16x32_f16(
                false, af.h, false, bf[t].h, (short)0, acc[t], false, false);

        if (k0 + 32 < K) {
            if (wave == 0) __builtin_amdgcn_s_wait_tensorcnt(0);
            __syncthreads();
            cur ^= 1;
        }
    }
#else
    for (int k0 = 0; k0 < K; k0 += 32) {
        Frag16 af;
        const _Float16* ap = Ab + (size_t)(i0 + lr) * lda + (k0 + lg * 16);
        af.u[0] = *(const v4u*)(ap);
        af.u[1] = *(const v4u*)(ap + 8);
        if (k0 + 32 < K) __builtin_prefetch(ap + 32, 0, 1);

        Frag16 bf[4];
#pragma unroll
        for (int t = 0; t < 4; ++t) {
            const _Float16* bp =
                Bb + (size_t)(j0 + t * 16 + lr) * ldb + (k0 + lg * 16);
            bf[t].u[0] = *(const v4u*)(bp);
            bf[t].u[1] = *(const v4u*)(bp + 8);
        }
#pragma unroll
        for (int t = 0; t < 4; ++t)
            acc[t] = __builtin_amdgcn_wmma_f32_16x16x32_f16(
                false, af.h, false, bf[t].h, (short)0, acc[t], false, false);
    }
#endif

    float*     Cb  = Co + (size_t)bz * sC;
    _Float16*  Chb = Ch ? (Ch + (size_t)bz * sC) : (_Float16*)0;
#pragma unroll
    for (int t = 0; t < 4; ++t) {
        const int j = j0 + t * 16 + lr;
        const float bv = bias ? bias[j] : 0.0f;
#pragma unroll
        for (int r = 0; r < 8; ++r) {
            const int row = i0 + lg * 8 + r;   // C/D ISA layout
            const size_t o = (size_t)row * ldc + j;
            float v = acc[t][r] + bv;
            if (accumulate) v += Cb[o];
            Cb[o] = v;
            if (Chb) Chb[o] = (_Float16)v;
        }
    }
}

// ===================== feature prep ==========================================
// x: [B,C,H*W] fp32 -> fh/fnh (f16 [B,N,C]), fT (f16 [B,C,N]), sq ([B,N])
__global__ void prep_kernel(const float* __restrict__ x,
                            _Float16* __restrict__ fh,
                            _Float16* __restrict__ fnh,
                            _Float16* __restrict__ fT,
                            float* __restrict__ sq)
{
    __shared__ float s[C_];
    __shared__ float stot;
    const int b = blockIdx.y, n = blockIdx.x, c = threadIdx.x;
    const float v = x[((size_t)b * C_ + c) * N_ + n];
    s[c] = v * v;
    __syncthreads();
    if (c == 0) {
        float t = 0.f;
        for (int k = 0; k < C_; ++k) t += s[k];
        stot = t;
    }
    __syncthreads();
    const float ss  = stot;
    const float nrm = fmaxf(sqrtf(ss), 1e-12f);
    const size_t o  = ((size_t)b * N_ + n) * C_ + c;
    fh[o]  = (_Float16)v;
    fnh[o] = (_Float16)(v / nrm);
    fT[((size_t)b * C_ + c) * N_ + n] = (_Float16)v;
    if (c == 0) sq[(size_t)b * N_ + n] = ss;
}

__global__ void cast_f16_kernel(const float* __restrict__ src,
                                _Float16* __restrict__ dst, int n)
{
    for (int i = blockIdx.x * blockDim.x + threadIdx.x; i < n;
         i += gridDim.x * blockDim.x)
        dst[i] = (_Float16)src[i];
}

__global__ void zero_kernel(float* __restrict__ p, int n)
{
    for (int i = blockIdx.x * blockDim.x + threadIdx.x; i < n;
         i += gridDim.x * blockDim.x)
        p[i] = 0.0f;
}

// ===================== distance from Gram ====================================
// mode 0: euclid  D = sqrt(max(sq_y[i] + sq_x[j] - 2G, 1e-12))
// mode 1: cosine  D = 1 - G   (G from normalized features)
__global__ void dist_kernel(const float* __restrict__ G,
                            const float* __restrict__ sqy,
                            const float* __restrict__ sqx,
                            float* __restrict__ D, int mode)
{
    const int total = B_ * NN2;
    for (int idx = blockIdx.x * blockDim.x + threadIdx.x; idx < total;
         idx += gridDim.x * blockDim.x) {
        const int b = idx >> 20, rem = idx & (NN2 - 1);
        const int i = rem >> 10, j = rem & (N_ - 1);
        float d;
        if (mode == 0) {
            float s = sqy[b * N_ + i] + sqx[b * N_ + j] - 2.0f * G[idx];
            d = sqrtf(fmaxf(s, 1e-12f));
        } else {
            d = 1.0f - G[idx];
        }
        D[idx] = d;
    }
}

// ===================== per-batch mean/std (two stage) ========================
__global__ void batch_partial_kernel(const float* __restrict__ D,
                                     float* __restrict__ psum,
                                     float* __restrict__ psumsq)
{
    __shared__ float s1[256], s2[256];
    const int b = blockIdx.y, blk = blockIdx.x, tid = threadIdx.x;
    const float* base = D + (size_t)b * NN2;
    float a = 0.f, q = 0.f;
    for (int t = blk * 256 + tid; t < NN2; t += 256 * 256) {
        const float v = base[t];
        a += v; q += v * v;
    }
    s1[tid] = a; s2[tid] = q;
    __syncthreads();
    for (int s = 128; s > 0; s >>= 1) {
        if (tid < s) { s1[tid] += s1[tid + s]; s2[tid] += s2[tid + s]; }
        __syncthreads();
    }
    if (tid == 0) {
        psum[b * 256 + blk]   = s1[0];
        psumsq[b * 256 + blk] = s2[0];
    }
}

__global__ void batch_final_kernel(const float* __restrict__ psum,
                                   const float* __restrict__ psumsq,
                                   float* __restrict__ mu,
                                   float* __restrict__ sigma)
{
    __shared__ float s1[256], s2[256];
    const int b = blockIdx.x, tid = threadIdx.x;
    s1[tid] = psum[b * 256 + tid];
    s2[tid] = psumsq[b * 256 + tid];
    __syncthreads();
    for (int s = 128; s > 0; s >>= 1) {
        if (tid < s) { s1[tid] += s1[tid + s]; s2[tid] += s2[tid + s]; }
        __syncthreads();
    }
    if (tid == 0) {
        const float M = (float)NN2;
        const float mean = s1[0] / M;
        const float var  = (s2[0] - s1[0] * s1[0] / M) / (M - 1.0f); // ddof=1
        mu[b] = mean;
        sigma[b] = sqrtf(fmaxf(var, 0.0f));
    }
}

// ===================== per-row stats: max, min, softmax-denom, entropy =======
__global__ void row_stats_kernel(const float* __restrict__ D,
                                 float* __restrict__ rmax,
                                 float* __restrict__ rmin,
                                 float* __restrict__ rsum,
                                 float* __restrict__ rent)
{
    __shared__ float ra[256], rb[256];
    const int b = blockIdx.y, i = blockIdx.x, tid = threadIdx.x;
    const float* drow = D + ((size_t)b * N_ + i) * N_;
    float mx = -3.0e38f, mn = 3.0e38f;
    for (int t = tid; t < N_; t += 256) {
        const float v = drow[t];
        mx = fmaxf(mx, v); mn = fminf(mn, v);
    }
    ra[tid] = mx; rb[tid] = mn;
    __syncthreads();
    for (int s = 128; s > 0; s >>= 1) {
        if (tid < s) {
            ra[tid] = fmaxf(ra[tid], ra[tid + s]);
            rb[tid] = fminf(rb[tid], rb[tid + s]);
        }
        __syncthreads();
    }
    const float dmax = ra[0], dmin = rb[0];
    __syncthreads();
    // softmax(-D/TAU): p = exp((dmin - d)/TAU)/sum
    float se = 0.f;
    for (int t = tid; t < N_; t += 256) se += __expf((dmin - drow[t]) * 0.1f);
    ra[tid] = se;
    __syncthreads();
    for (int s = 128; s > 0; s >>= 1) {
        if (tid < s) ra[tid] += ra[tid + s];
        __syncthreads();
    }
    const float S = ra[0];
    __syncthreads();
    const float inv = 1.0f / S;
    float ent = 0.f;
    for (int t = tid; t < N_; t += 256) {
        const float p = __expf((dmin - drow[t]) * 0.1f) * inv;
        ent -= p * __logf(p + 1e-8f);
    }
    rb[tid] = ent;
    __syncthreads();
    for (int s = 128; s > 0; s >>= 1) {
        if (tid < s) rb[tid] += rb[tid + s];
        __syncthreads();
    }
    if (tid == 0) {
        const size_t o = (size_t)b * N_ + i;
        rmax[o] = dmax; rmin[o] = dmin; rsum[o] = S; rent[o] = rb[0];
    }
}

// ===================== per-row sort + cumsum threshold -> adjacency ==========
__global__ void sort_threshold_kernel(const float* __restrict__ D,
                                      const float* __restrict__ rmin,
                                      const float* __restrict__ rsum,
                                      const float* __restrict__ rmax,
                                      const float* __restrict__ rent,
                                      const float* __restrict__ mu,
                                      const float* __restrict__ sigma,
                                      unsigned char* __restrict__ adj)
{
    __shared__ float skey[N_];
    __shared__ int   sidx[N_];
    __shared__ float sred[256];
    __shared__ int   sri[256];
    const int b = blockIdx.y, i = blockIdx.x, tid = threadIdx.x;
    const size_t rowo = (size_t)b * N_ + i;
    const float* drow = D + rowo * N_;

    for (int t = tid; t < N_; t += 256) { skey[t] = drow[t]; sidx[t] = t; }
    __syncthreads();

    // bitonic sort ascending (1024 elems, 256 threads)
    for (int k = 2; k <= N_; k <<= 1) {
        for (int j = k >> 1; j > 0; j >>= 1) {
            for (int t = tid; t < N_; t += 256) {
                const int ixj = t ^ j;
                if (ixj > t) {
                    const bool up = ((t & k) == 0);
                    const float a = skey[t], c = skey[ixj];
                    if ((a > c) == up) {
                        skey[t] = c; skey[ixj] = a;
                        const int ia = sidx[t];
                        sidx[t] = sidx[ixj]; sidx[ixj] = ia;
                    }
                }
            }
            __syncthreads();
        }
    }

    // inclusive cumsum of sorted probs (blocked: 4 elems/thread)
    const float dmin = rmin[rowo], inv = 1.0f / rsum[rowo];
    const int base = tid * 4;
    float lp[4]; float run = 0.f;
#pragma unroll
    for (int q = 0; q < 4; ++q) {
        const float p = __expf((dmin - skey[base + q]) * 0.1f) * inv;
        run += p; lp[q] = run;
    }
    sred[tid] = run;
    __syncthreads();
    for (int s = 1; s < 256; s <<= 1) {
        const float v = (tid >= s) ? sred[tid - s] : 0.0f;
        __syncthreads();
        sred[tid] += v;
        __syncthreads();
    }
    const float off = sred[tid] - run;

    // first sorted position with cum > 0.8 (min-reduce), else N_
    int firstLocal = N_;
#pragma unroll
    for (int q = 0; q < 4; ++q)
        if (off + lp[q] > 0.8f) { firstLocal = base + q; break; }
    sri[tid] = firstLocal;
    __syncthreads();
    for (int s = 128; s > 0; s >>= 1) {
        if (tid < s) sri[tid] = min(sri[tid], sri[tid + s]);
        __syncthreads();
    }
    const int num_edges = sri[0];

    // adj[i][j] = (rank(j) < num_edges) && (D < T)
    const float thbase = mu[b] + 0.08f * sigma[b];     // ALPHA
    const float dmax  = rmax[rowo];
    const float decay = __expf(-rent[rowo]);           // ETA = 1
    unsigned char* arow = adj + rowo * N_;
#pragma unroll
    for (int q = 0; q < 4; ++q) {
        const int r = base + q;
        const float d = skey[r];
        const float T = thbase + 0.01f * (1.0f - d / dmax) * decay;  // BETA
        arow[sidx[r]] = (r < num_edges && d < T) ? (unsigned char)1
                                                 : (unsigned char)0;
    }
}

// ===================== masked row softmax -> f16 attention ===================
__global__ void masked_softmax_kernel(const float* __restrict__ L,
                                      const unsigned char* __restrict__ A,
                                      _Float16* __restrict__ att)
{
    __shared__ float sr[256];
    const int b = blockIdx.y, i = blockIdx.x, tid = threadIdx.x;
    const size_t rowo = ((size_t)b * N_ + i) * N_;
    const float* lrow = L + rowo;
    const unsigned char* arow = A + rowo;

    float mx = -3.0e38f;
    for (int t = tid; t < N_; t += 256) {
        const float v = arow[t] ? lrow[t] : 0.0f;
        mx = fmaxf(mx, v);
    }
    sr[tid] = mx;
    __syncthreads();
    for (int s = 128; s > 0; s >>= 1) {
        if (tid < s) sr[tid] = fmaxf(sr[tid], sr[tid + s]);
        __syncthreads();
    }
    const float m = sr[0];
    __syncthreads();
    float se = 0.f;
    for (int t = tid; t < N_; t += 256) {
        const float v = arow[t] ? lrow[t] : 0.0f;
        se += __expf(v - m);
    }
    sr[tid] = se;
    __syncthreads();
    for (int s = 128; s > 0; s >>= 1) {
        if (tid < s) sr[tid] += sr[tid + s];
        __syncthreads();
    }
    const float inv = 1.0f / sr[0];
    for (int t = tid; t < N_; t += 256) {
        const float v = arow[t] ? lrow[t] : 0.0f;
        att[rowo + t] = (_Float16)(__expf(v - m) * inv);
    }
}

// ===================== O [B,N,C] -> out [B,C,H,W] ============================
__global__ void out_kernel(const float* __restrict__ O, float* __restrict__ out)
{
    const int total = B_ * C_ * N_;
    for (int idx = blockIdx.x * blockDim.x + threadIdx.x; idx < total;
         idx += gridDim.x * blockDim.x) {
        const int b = idx >> 16, rem = idx & 65535;
        const int c = rem >> 10, n = rem & (N_ - 1);
        out[idx] = O[((size_t)b * N_ + n) * C_ + c];
    }
}

// ===================== host orchestration ====================================
extern "C" void kernel_launch(void* const* d_in, const int* in_sizes, int n_in,
                              void* d_out, int out_size, void* d_ws,
                              size_t ws_size, hipStream_t stream)
{
    (void)in_sizes; (void)n_in; (void)out_size; (void)ws_size;

    const float* x[3]  = {(const float*)d_in[0], (const float*)d_in[1],
                          (const float*)d_in[2]};
    const float* w[3]  = {(const float*)d_in[3], (const float*)d_in[5],
                          (const float*)d_in[7]};
    const float* bs[3] = {(const float*)d_in[4], (const float*)d_in[6],
                          (const float*)d_in[8]};

    char* ws = (char*)d_ws;
    size_t off = 0;
    auto carve = [&](size_t bytes) -> char* {
        char* p = ws + off;
        off += (bytes + 255) & ~(size_t)255;
        return p;
    };

    const size_t FSZ  = (size_t)B_ * N_ * C_;   // per-feature elems
    _Float16 *fh[3], *fnh[3], *fT[3], *ph[3], *wh[3];
    float *pf[3], *sq[3];
    for (int s = 0; s < 3; ++s) fh[s]  = (_Float16*)carve(FSZ * 2);
    for (int s = 0; s < 3; ++s) fnh[s] = (_Float16*)carve(FSZ * 2);
    for (int s = 0; s < 3; ++s) fT[s]  = (_Float16*)carve(FSZ * 2);
    for (int s = 0; s < 3; ++s) ph[s]  = (_Float16*)carve(FSZ * 2);
    for (int s = 0; s < 3; ++s) wh[s]  = (_Float16*)carve((size_t)C_ * C_ * 2);
    for (int s = 0; s < 3; ++s) pf[s]  = (float*)carve(FSZ * 4);
    for (int s = 0; s < 3; ++s) sq[s]  = (float*)carve((size_t)B_ * N_ * 4);

    float* G = (float*)carve((size_t)B_ * NN2 * 4);
    float* D = (float*)carve((size_t)B_ * NN2 * 4);
    unsigned char* adj[9];
    for (int s = 0; s < 9; ++s) adj[s] = (unsigned char*)carve((size_t)B_ * NN2);
    _Float16* atth = (_Float16*)carve((size_t)B_ * NN2 * 2);
    float* rmax = (float*)carve((size_t)B_ * N_ * 4);
    float* rmin = (float*)carve((size_t)B_ * N_ * 4);
    float* rsum = (float*)carve((size_t)B_ * N_ * 4);
    float* rent = (float*)carve((size_t)B_ * N_ * 4);
    float* psum   = (float*)carve((size_t)B_ * 256 * 4);
    float* psumsq = (float*)carve((size_t)B_ * 256 * 4);
    float* mu    = (float*)carve((size_t)B_ * 4);
    float* sigma = (float*)carve((size_t)B_ * 4);
    float* O = (float*)carve(FSZ * 4);

    const long sF = (long)N_ * C_;   // per-batch stride, feature mats
    const long sG = (long)NN2;       // per-batch stride, NxN mats

    auto gemm = [&](const _Float16* A, int lda, long sA, const _Float16* Bm,
                    int ldb, long sB, float* Co, int ldc, long sC,
                    _Float16* Ch, const float* bias, int M, int Nn, int K,
                    int accum) {
        dim3 g(Nn / 64, M / 64, B_), blk(128);
        wmma_gemm_nt<<<g, blk, 0, stream>>>(A, lda, sA, Bm, ldb, sB, Co, ldc,
                                            sC, Ch, bias, K, accum);
    };

    // --- features + norms ---------------------------------------------------
    for (int s = 0; s < 3; ++s)
        prep_kernel<<<dim3(N_, B_), C_, 0, stream>>>(x[s], fh[s], fnh[s],
                                                     fT[s], sq[s]);
    for (int s = 0; s < 3; ++s)
        cast_f16_kernel<<<16, 256, 0, stream>>>(w[s], wh[s], C_ * C_);

    // --- p_s = f_s * w_s^T + b_s  (M=1024, N=64, K=64), f16 mirror for WMMA --
    for (int s = 0; s < 3; ++s)
        gemm(fh[s], C_, sF, wh[s], C_, 0, pf[s], C_, sF, ph[s], bs[s],
             N_, C_, C_, 0);

    // --- 9 similarity blocks -> adjacency masks -----------------------------
    struct PairT { int y, xx, mode, slot; };
    const PairT pairs[9] = {
        {0, 0, 0, 0}, {1, 1, 0, 1}, {2, 2, 0, 2},          // euclid self
        {0, 1, 1, 3}, {1, 0, 1, 4}, {0, 2, 1, 5},          // cosine cross
        {2, 0, 1, 6}, {1, 2, 1, 7}, {2, 1, 1, 8}};
    for (int pi = 0; pi < 9; ++pi) {
        const PairT& P = pairs[pi];
        const _Float16* Ay = (P.mode == 0) ? fh[P.y] : fnh[P.y];
        const _Float16* Bx = (P.mode == 0) ? fh[P.xx] : fnh[P.xx];
        gemm(Ay, C_, sF, Bx, C_, sF, G, N_, sG, 0, 0, N_, N_, C_, 0);
        dist_kernel<<<2048, 256, 0, stream>>>(G, sq[P.y], sq[P.xx], D, P.mode);
        batch_partial_kernel<<<dim3(256, B_), 256, 0, stream>>>(D, psum, psumsq);
        batch_final_kernel<<<B_, 256, 0, stream>>>(psum, psumsq, mu, sigma);
        row_stats_kernel<<<dim3(N_, B_), 256, 0, stream>>>(D, rmax, rmin, rsum,
                                                           rent);
        sort_threshold_kernel<<<dim3(N_, B_), 256, 0, stream>>>(
            D, rmin, rsum, rmax, rent, mu, sigma, adj[P.slot]);
    }

    // --- attention + aggregation: O = sum_{k,j} softmax(A_kj . p_k p_j^T) f_j
    zero_kernel<<<512, 256, 0, stream>>>(O, (int)FSZ);
    const int slotKJ[3][3] = {{0, 3, 5}, {4, 1, 7}, {6, 8, 2}};
    for (int k = 0; k < 3; ++k) {
        for (int j = 0; j < 3; ++j) {
            gemm(ph[k], C_, sF, ph[j], C_, sF, G, N_, sG, 0, 0, N_, N_, C_, 0);
            masked_softmax_kernel<<<dim3(N_, B_), 256, 0, stream>>>(
                G, adj[slotKJ[k][j]], atth);
            gemm(atth, N_, sG, fT[j], N_, sF, O, C_, sF, 0, 0, N_, C_, N_, 1);
        }
    }

    out_kernel<<<512, 256, 0, stream>>>(O, (float*)d_out);
}
